// MambaPhysiNet_11158325035370
// MI455X (gfx1250) — matrix operand order repaired and checked by reading, not compiled, not run on hardware
//
#include <hip/hip_runtime.h>
#include <stddef.h>
#include <stdint.h>

// ---------------------------------------------------------------------------
// Types for CDNA5 WMMA (wave32, 16x16x32 bf16 -> f32)
// ---------------------------------------------------------------------------
typedef __bf16 bf16;
typedef __bf16 v16bf __attribute__((ext_vector_type(16)));
typedef __bf16 v8bf  __attribute__((ext_vector_type(8)));
typedef float  v8f   __attribute__((ext_vector_type(8)));
typedef unsigned int v4u __attribute__((ext_vector_type(4)));
typedef int v8i __attribute__((ext_vector_type(8)));
typedef int v4i __attribute__((ext_vector_type(4)));

#define EPSBN 1e-5f

#if defined(__has_builtin)
#if __has_builtin(__builtin_amdgcn_tensor_load_to_lds) && \
    __has_builtin(__builtin_amdgcn_s_wait_tensorcnt)
#define HAVE_TDM 1
#endif
#endif
#ifndef HAVE_TDM
#define HAVE_TDM 0
#endif

// ---------------------------------------------------------------------------
// WMMA fragment mapping (ISA 7.12.2), wave32:
//   A 16x32 bf16 : m = lane%16 ; elem j<8 -> K = 8*(lane/16)+j
//                                elem j>=8 -> K = 16 + 8*(lane/16)+(j-8)
//   B 32x16 bf16 : n = lane%16 ; same K mapping as A            (Bt[n][k])
//   C 16x16 f32  : n = lane%16 ; vgpr r -> m = r + 8*(lane/16)
// ---------------------------------------------------------------------------

// --------------------------- generic bf16 GEMM -----------------------------
// Grid: blockIdx.y = M-tile (shared by all 8 waves), blockIdx.x covers groups
// of 8 N-tiles. A 16xKp panel staged to LDS once per block via the Tensor
// Data Mover (TDM); B streams from global (L2 resident).
// Requires: M%16==0, Kp%32==0, Kp<=512, Bt has Npad=ceil16(N) zero-padded rows.
__global__ __launch_bounds__(256)
void gemm_wmma_bf16(const bf16* __restrict__ A, const bf16* __restrict__ Bt,
                    float* __restrict__ C, const float* __restrict__ bias,
                    int M, int N, int Kp, int ldc) {
  int lane = (int)(threadIdx.x & 31u);
  int wv   = (int)(threadIdx.x >> 5);
  int tilesN = (N + 15) >> 4;
  int tm = blockIdx.y;
  int tn = blockIdx.x * 8 + wv;
  int l16 = lane & 15;
  int half = lane >> 4;

#if HAVE_TDM
  __shared__ bf16 As[16 * 512];
  if (wv == 0) {
    // ---- Tensor DMA: load A[tm*16 .. +16), [0..Kp) into LDS (packed) ----
    uint64_t ga = (uint64_t)(uintptr_t)(A + (size_t)tm * 16 * Kp);
    unsigned ldsoff = (unsigned)(uintptr_t)(&As[0]);  // low 32 bits = LDS byte offset
    v4u g0;
    g0[0] = 1u;                                   // count=1, user mode
    g0[1] = ldsoff;                               // lds_addr
    g0[2] = (unsigned)ga;                         // global_addr[31:0]
    g0[3] = (unsigned)((ga >> 32) & 0x01FFFFFFu) | (2u << 30);  // [56:32] | type=2
    unsigned uKp = (unsigned)Kp;
    v8i g1;
    g1[0] = (int)(1u << 16);                      // wg_mask=0, data_size=1 (2B)
    g1[1] = (int)((uKp & 0xFFFFu) << 16);         // tensor_dim0[15:0]
    g1[2] = (int)((uKp >> 16) | (16u << 16));     // dim0[31:16] | tensor_dim1[15:0]=16
    g1[3] = (int)(uKp << 16);                     // dim1[31:16]=0 | tile_dim0=Kp
    g1[4] = 16;                                   // tile_dim1=16 | tile_dim2=0
    g1[5] = (int)uKp;                             // tensor_dim0_stride[31:0]
    g1[6] = 0;                                    // stride0[47:32] | stride1[15:0]
    g1[7] = 0;                                    // stride1[47:32]
    v4i z4 = {0, 0, 0, 0};
    v8i z8 = {0, 0, 0, 0, 0, 0, 0, 0};
    __builtin_amdgcn_tensor_load_to_lds(g0, g1, z4, z4, z8, 0);
    __builtin_amdgcn_s_wait_tensorcnt(0);
  }
  __syncthreads();
  const bf16* arow = As + (size_t)l16 * Kp;
#else
  const bf16* arow = A + ((size_t)(tm * 16 + l16)) * Kp;
#endif

  if (tn >= tilesN || tm * 16 >= M) return;  // after barrier; wave-uniform
  int n = tn * 16 + l16;
  const bf16* brow = Bt + (size_t)n * Kp;    // Npad rows, zero padded
  v8f acc = {};
  for (int k0 = 0; k0 < Kp; k0 += 32) {
    v8bf a0 = *(const v8bf*)(arow + k0 + half * 8);
    v8bf a1 = *(const v8bf*)(arow + k0 + 16 + half * 8);
    v8bf b0 = *(const v8bf*)(brow + k0 + half * 8);
    v8bf b1 = *(const v8bf*)(brow + k0 + 16 + half * 8);
    v16bf av, bv;
#pragma unroll
    for (int i = 0; i < 8; ++i) {
      av[i] = a0[i]; av[i + 8] = a1[i];
      bv[i] = b0[i]; bv[i + 8] = b1[i];
    }
    acc = __builtin_amdgcn_wmma_f32_16x16x32_bf16(false, av, false, bv,
                                                  (short)0, acc, false, false);
  }
  if (n < N) {
    float bb = bias ? bias[n] : 0.f;
#pragma unroll
    for (int r = 0; r < 8; ++r) {
      int mm = tm * 16 + r + half * 8;
      C[(size_t)mm * ldc + n] = acc[r] + bb;
    }
  }
}

// ----------------- implicit-im2col 3x3 SAME conv via WMMA ------------------
// in: NHWC f32 [Bn,H,W,Cin]; out: [Bn*H*W, Cout] f32 (== NHWC).
// chanMajor==1 (requires Cin%32==0): K ordered k=(ky*3+kx)*Cin+c -> the
// A-fragment is 4 contiguous float4 loads per K-step (vectorized im2col).
// chanMajor==0: K ordered k=c*9+(ky*3+kx), scalar gather (used for Cin=3).
// Bt must be packed with the matching K order.
__global__ __launch_bounds__(256)
void conv3x3_wmma(const float* __restrict__ in, const bf16* __restrict__ Bt,
                  float* __restrict__ out, const float* __restrict__ bias,
                  int Bn, int H, int W, int Cin, int Cout, int chanMajor) {
  int wave = (int)((blockIdx.x * blockDim.x + threadIdx.x) >> 5);
  int lane = (int)(threadIdx.x & 31u);
  int M = Bn * H * W;
  int tilesN = (Cout + 15) >> 4;
  int tm = wave / tilesN, tn = wave % tilesN;
  if (tm * 16 >= M) return;
  int l16 = lane & 15, half = lane >> 4;
  int m = tm * 16 + l16;
  int b = m / (H * W);
  int rem = m % (H * W);
  int y = rem / W, x = rem % W;
  int Kdim = Cin * 9;
  int Kp = (Kdim + 31) & ~31;
  int n = tn * 16 + l16;
  const bf16* brow = Bt + (size_t)n * Kp;
  v8f acc = {};
  for (int k0 = 0; k0 < Kdim; k0 += 32) {
    v16bf av;
    if (chanMajor) {
      int r9 = k0 / Cin;              // constant across this 32-block (Cin%32==0)
      int cbase = k0 - r9 * Cin;
      int ky = r9 / 3, kx = r9 - ky * 3;
      int yy = y + ky - 1, xx = x + kx - 1;
      if (yy >= 0 && yy < H && xx >= 0 && xx < W) {
        const float* p = in + (((size_t)b * H + yy) * W + xx) * Cin + cbase;
        float4 lo0 = *(const float4*)(p + half * 8);
        float4 lo1 = *(const float4*)(p + half * 8 + 4);
        float4 hi0 = *(const float4*)(p + 16 + half * 8);
        float4 hi1 = *(const float4*)(p + 16 + half * 8 + 4);
        av[0] = (bf16)lo0.x; av[1] = (bf16)lo0.y; av[2] = (bf16)lo0.z; av[3] = (bf16)lo0.w;
        av[4] = (bf16)lo1.x; av[5] = (bf16)lo1.y; av[6] = (bf16)lo1.z; av[7] = (bf16)lo1.w;
        av[8] = (bf16)hi0.x; av[9] = (bf16)hi0.y; av[10] = (bf16)hi0.z; av[11] = (bf16)hi0.w;
        av[12] = (bf16)hi1.x; av[13] = (bf16)hi1.y; av[14] = (bf16)hi1.z; av[15] = (bf16)hi1.w;
      } else {
#pragma unroll
        for (int j = 0; j < 16; ++j) av[j] = (bf16)0.f;
      }
    } else {
#pragma unroll
      for (int j = 0; j < 16; ++j) {
        int k = k0 + ((j < 8) ? (half * 8 + j) : (16 + half * 8 + (j - 8)));
        float v = 0.f;
        if (k < Kdim) {
          int c = k / 9, r9 = k % 9;
          int ky = r9 / 3, kx = r9 % 3;
          int yy = y + ky - 1, xx = x + kx - 1;
          if (yy >= 0 && yy < H && xx >= 0 && xx < W)
            v = in[(((size_t)b * H + yy) * W + xx) * Cin + c];
        }
        av[j] = (bf16)v;
      }
    }
    v8bf b0 = *(const v8bf*)(brow + k0 + half * 8);
    v8bf b1 = *(const v8bf*)(brow + k0 + 16 + half * 8);
    v16bf bv;
#pragma unroll
    for (int i = 0; i < 8; ++i) { bv[i] = b0[i]; bv[i + 8] = b1[i]; }
    acc = __builtin_amdgcn_wmma_f32_16x16x32_bf16(false, av, false, bv,
                                                  (short)0, acc, false, false);
  }
  if (n < Cout) {
    float bb = bias ? bias[n] : 0.f;
#pragma unroll
    for (int r = 0; r < 8; ++r) {
      int mm = tm * 16 + r + half * 8;
      out[(size_t)mm * Cout + n] = acc[r] + bb;
    }
  }
}

// ----------------------- weight pack: f32 -> Bt bf16 -----------------------
// mode 0: direct      Bt[n][k] = src[n*K + k]
// mode 1: transpose   Bt[n][k] = src[k*N + n]
// mode 2: conv chan-major: k = (ky*3+kx)*Cin + c ; src = W[n][c][ky][kx]
__global__ void prep_bt(const float* __restrict__ src, bf16* __restrict__ dst,
                        int K, int N, int Kp, int Np, int mode, int Cin) {
  size_t i = (size_t)blockIdx.x * blockDim.x + threadIdx.x;
  size_t total = (size_t)Np * Kp;
  if (i >= total) return;
  int n = (int)(i / Kp), k = (int)(i % Kp);
  float v = 0.f;
  if (n < N && k < K) {
    if (mode == 0)      v = src[(size_t)n * K + k];
    else if (mode == 1) v = src[(size_t)k * N + n];
    else {
      int r9 = k / Cin, c = k - r9 * Cin;
      v = src[((size_t)n * Cin + c) * 9 + r9];
    }
  }
  dst[i] = (bf16)v;
}

// ----------------------- activation f32 -> bf16 (padded) -------------------
__global__ void cvt_bf16(const float* __restrict__ src, bf16* __restrict__ dst,
                         int M, int Kcols, int src_ld, int col0, int Kp) {
  size_t i = (size_t)blockIdx.x * blockDim.x + threadIdx.x;
  if (i >= (size_t)M * Kp) return;
  int k = (int)(i % Kp);
  size_t m = i / Kp;
  float v = (k < Kcols) ? src[m * (size_t)src_ld + col0 + k] : 0.f;
  dst[i] = (bf16)v;
}

// ------------------------------ BatchNorm ----------------------------------
__global__ void bn_stats(const float* __restrict__ x, int M, int C,
                         float* __restrict__ mean, float* __restrict__ var) {
  int c = blockIdx.x;
  float s = 0.f, ss = 0.f;
  for (int i = threadIdx.x; i < M; i += blockDim.x) {
    float v = x[(size_t)i * C + c];
    s += v; ss += v * v;
  }
  __shared__ float sh[256], sh2[256];
  sh[threadIdx.x] = s; sh2[threadIdx.x] = ss;
  __syncthreads();
  for (int o = blockDim.x >> 1; o > 0; o >>= 1) {
    if ((int)threadIdx.x < o) {
      sh[threadIdx.x] += sh[threadIdx.x + o];
      sh2[threadIdx.x] += sh2[threadIdx.x + o];
    }
    __syncthreads();
  }
  if (threadIdx.x == 0) {
    float m = sh[0] / (float)M;
    mean[c] = m;
    var[c] = sh2[0] / (float)M - m * m;
  }
}

// act: 0=lrelu, 1=relu, 2=sigmoid, 3=none
__global__ void bn_apply(float* __restrict__ x, const float* __restrict__ mean,
                         const float* __restrict__ var, const float* __restrict__ g,
                         const float* __restrict__ bt, int M, int C, int act) {
  size_t i = (size_t)blockIdx.x * blockDim.x + threadIdx.x;
  if (i >= (size_t)M * C) return;
  int c = (int)(i % C);
  float v = x[i];
  v = (v - mean[c]) * rsqrtf(var[c] + EPSBN) * g[c] + bt[c];
  if (act == 0)      v = (v >= 0.f) ? v : 0.2f * v;
  else if (act == 1) v = fmaxf(v, 0.f);
  else if (act == 2) v = 1.f / (1.f + expf(-v));
  x[i] = v;
}

__global__ void act_only(float* __restrict__ x, size_t n, int act) {
  size_t i = (size_t)blockIdx.x * blockDim.x + threadIdx.x;
  if (i >= n) return;
  float v = x[i];
  if (act == 0)      v = (v >= 0.f) ? v : 0.2f * v;
  else if (act == 1) v = fmaxf(v, 0.f);
  x[i] = v;
}

// --------------------------- pool / resample / concat ----------------------
__global__ void maxpool2_nhwc(const float* __restrict__ in, float* __restrict__ out,
                              int Bn, int H, int W, int C) {
  size_t i = (size_t)blockIdx.x * blockDim.x + threadIdx.x;
  size_t total = (size_t)Bn * (H / 2) * (W / 2) * C;
  if (i >= total) return;
  int c = (int)(i % C);
  size_t t = i / C;
  int x = (int)(t % (W / 2)); t /= (W / 2);
  int y = (int)(t % (H / 2)); int b = (int)(t / (H / 2));
  const float* p = in + (((size_t)b * H + 2 * y) * W + 2 * x) * C + c;
  size_t rs = (size_t)W * C;
  float v = fmaxf(fmaxf(p[0], p[C]), fmaxf(p[rs], p[rs + C]));
  out[i] = v;
}

__global__ void upsample2_nhwc(const float* __restrict__ in, float* __restrict__ out,
                               int Bn, int H, int W, int C) {
  int OH = 2 * H, OW = 2 * W;
  size_t i = (size_t)blockIdx.x * blockDim.x + threadIdx.x;
  size_t total = (size_t)Bn * OH * OW * C;
  if (i >= total) return;
  int c = (int)(i % C);
  size_t t = i / C;
  int ox = (int)(t % OW); t /= OW;
  int oy = (int)(t % OH); int b = (int)(t / OH);
  float py = (float)oy * (float)(H - 1) / (float)(OH - 1);
  float px = (float)ox * (float)(W - 1) / (float)(OW - 1);
  int y0 = (int)floorf(py), x0 = (int)floorf(px);
  int y1 = min(y0 + 1, H - 1), x1 = min(x0 + 1, W - 1);
  float wy = py - (float)y0, wx = px - (float)x0;
  const float* base = in + (size_t)b * H * W * C + c;
  float v00 = base[((size_t)y0 * W + x0) * C];
  float v01 = base[((size_t)y0 * W + x1) * C];
  float v10 = base[((size_t)y1 * W + x0) * C];
  float v11 = base[((size_t)y1 * W + x1) * C];
  out[i] = (1.f - wy) * ((1.f - wx) * v00 + wx * v01) +
           wy * ((1.f - wx) * v10 + wx * v11);
}

__global__ void concat_c(const float* __restrict__ a, const float* __restrict__ b,
                         float* __restrict__ out, size_t Mpos, int Ca, int Cb) {
  int C = Ca + Cb;
  size_t i = (size_t)blockIdx.x * blockDim.x + threadIdx.x;
  if (i >= Mpos * (size_t)C) return;
  int c = (int)(i % C);
  size_t p = i / C;
  out[i] = (c < Ca) ? a[p * Ca + c] : b[p * Cb + (c - Ca)];
}

// ------------------------------ layout swaps -------------------------------
__global__ void nchw2nhwc(const float* __restrict__ in, float* __restrict__ out,
                          int Bn, int C, int H, int W) {
  size_t i = (size_t)blockIdx.x * blockDim.x + threadIdx.x;
  if (i >= (size_t)Bn * C * H * W) return;
  int c = (int)(i % C); size_t t = i / C;
  int x = (int)(t % W); t /= W;
  int y = (int)(t % H); int b = (int)(t / H);
  out[i] = in[(((size_t)b * C + c) * H + y) * W + x];
}

__global__ void nhwc2nchw(const float* __restrict__ in, float* __restrict__ out,
                          int Bn, int C, int H, int W) {
  size_t i = (size_t)blockIdx.x * blockDim.x + threadIdx.x;
  if (i >= (size_t)Bn * C * H * W) return;
  int x = (int)(i % W); size_t t = i / W;
  int y = (int)(t % H); t /= H;
  int c = (int)(t % C); int b = (int)(t / C);
  out[i] = in[(((size_t)b * H + y) * W + x) * C + c];
}

// ------------------------------ Mamba pieces -------------------------------
__global__ void gate_softmax(const float* __restrict__ seq, const float* __restrict__ gw,
                             const float* __restrict__ gb, float* __restrict__ gate,
                             int Mrows, int Cc) {
  int m = (int)(blockIdx.x * blockDim.x + threadIdx.x);
  if (m >= Mrows) return;
  float a0 = gb[0], a1 = gb[1], a2 = gb[2];
  const float* row = seq + (size_t)m * Cc;
  for (int c = 0; c < Cc; ++c) {
    float v = row[c];
    a0 += v * gw[c * 3 + 0];
    a1 += v * gw[c * 3 + 1];
    a2 += v * gw[c * 3 + 2];
  }
  float mx = fmaxf(a0, fmaxf(a1, a2));
  float e0 = expf(a0 - mx), e1 = expf(a1 - mx), e2 = expf(a2 - mx);
  float s = e0 + e1 + e2;
  gate[(size_t)m * 3 + 0] = e0 / s;
  gate[(size_t)m * 3 + 1] = e1 / s;
  gate[(size_t)m * 3 + 2] = e2 / s;
}

// causal depthwise conv (K=4, left pad 3) + SiLU; xz [B*L, 2*DI] -> xi [B*L, DI]
__global__ void dwconv_silu(const float* __restrict__ xz, const float* __restrict__ w,
                            const float* __restrict__ bconv, float* __restrict__ xi,
                            int Bn, int L, int DIc) {
  size_t i = (size_t)blockIdx.x * blockDim.x + threadIdx.x;
  if (i >= (size_t)Bn * L * DIc) return;
  int d = (int)(i % DIc);
  size_t t = i / DIc;
  int l = (int)(t % L); int b = (int)(t / L);
  float acc = bconv[d];
#pragma unroll
  for (int tt = 0; tt < 4; ++tt) {
    int ls = l - 3 + tt;
    if (ls >= 0)
      acc += w[d * 4 + tt] * xz[((size_t)b * L + ls) * (2 * DIc) + d];
  }
  xi[i] = acc / (1.f + expf(-acc));
}

__global__ void softplus_k(float* __restrict__ x, size_t n) {
  size_t i = (size_t)blockIdx.x * blockDim.x + threadIdx.x;
  if (i >= n) return;
  float v = x[i];
  x[i] = (v > 20.f) ? v : log1pf(expf(v));
}

// selective SSM scan; one thread per (b,d); 16-wide state kept in registers.
__global__ void ssm_scan(const float* __restrict__ delta, const float* __restrict__ dbl,
                         const float* __restrict__ xi, const float* __restrict__ xz,
                         const float* __restrict__ A_log, const float* __restrict__ Dp,
                         float* __restrict__ y, int Bn, int L, int DIc) {
  int t = (int)(blockIdx.x * blockDim.x + threadIdx.x);
  if (t >= Bn * DIc) return;
  int b = t / DIc, d = t % DIc;
  float A[16], h[16];
#pragma unroll
  for (int s = 0; s < 16; ++s) {
    A[s] = -expf(A_log[d * 16 + s]);
    h[s] = 0.f;
  }
  float Dd = Dp[d];
  for (int l = 0; l < L; ++l) {
    size_t row = (size_t)b * L + l;
    float dl = delta[row * DIc + d];
    float xv = xi[row * DIc + d];
    const float* Bm = dbl + row * 48 + 16;
    const float* Cm = dbl + row * 48 + 32;
    float acc = 0.f;
#pragma unroll
    for (int s = 0; s < 16; ++s) {
      float dA = expf(dl * A[s]);
      float h2 = dA * h[s] + dl * Bm[s] * xv;
      h[s] = h2;
      acc += h2 * Cm[s];
    }
    float yv = acc + Dd * xv;
    float zv = xz[row * (size_t)(2 * DIc) + DIc + d];
    yv *= zv / (1.f + expf(-zv));
    y[row * DIc + d] = yv;
  }
}

__global__ void gate_accum(float* __restrict__ mix, const float* __restrict__ tmp,
                           const float* __restrict__ gate, int e, size_t Mrows,
                           int Cc, int init) {
  size_t i = (size_t)blockIdx.x * blockDim.x + threadIdx.x;
  if (i >= Mrows * (size_t)Cc) return;
  size_t m = i / Cc;
  float v = gate[m * 3 + e] * tmp[i];
  mix[i] = init ? v : (mix[i] + v);
}

// --------------- 100-step heat diffusion, per-(b,c) plane in LDS -----------
__global__ __launch_bounds__(256)
void diffusion_k(float* __restrict__ h, int Bn, int C, int steps) {
  int bc = blockIdx.x;
  int b = bc / C, c = bc % C;
  __shared__ float s0[1024], s1[1024];
  float* cur = s0;
  float* nxt = s1;
  for (int i = threadIdx.x; i < 1024; i += blockDim.x)
    s0[i] = h[(((size_t)b * 32 + (i >> 5)) * 32 + (i & 31)) * C + c];
  __syncthreads();
  for (int st = 0; st < steps; ++st) {
    for (int i = threadIdx.x; i < 1024; i += blockDim.x) {
      int y = i >> 5, x = i & 31;
      float v = cur[i];
      float lap = -4.f * v;
      if (y > 0)  lap += cur[i - 32];
      if (y < 31) lap += cur[i + 32];
      if (x > 0)  lap += cur[i - 1];
      if (x < 31) lap += cur[i + 1];
      nxt[i] = v + 0.1f * (1.0f * lap - 0.5f * v);
    }
    __syncthreads();
    float* tp = cur; cur = nxt; nxt = tp;
  }
  for (int i = threadIdx.x; i < 1024; i += blockDim.x)
    h[(((size_t)b * 32 + (i >> 5)) * 32 + (i & 31)) * C + c] = cur[i];
}

// ---------------------------------------------------------------------------
// Host orchestration
// ---------------------------------------------------------------------------
static inline unsigned gblk(size_t n) { return (unsigned)((n + 255) / 256); }

static inline void run_gemm(const bf16* A, const bf16* Bt, float* C,
                            const float* bias, int M, int N, int Kp, int ldc,
                            hipStream_t s) {
  int tilesN = (N + 15) / 16;
  dim3 grid((unsigned)((tilesN + 7) / 8), (unsigned)(M / 16));
  gemm_wmma_bf16<<<grid, 256, 0, s>>>(A, Bt, C, bias, M, N, Kp, ldc);
}

static inline void run_conv(const float* in, const bf16* Bt, float* out,
                            const float* bias, int Bn, int H, int W, int Cin,
                            int Cout, int chanMajor, hipStream_t s) {
  int M = Bn * H * W;
  int tiles = (M / 16) * ((Cout + 15) / 16);
  int blocks = (tiles + 7) / 8;
  conv3x3_wmma<<<blocks, 256, 0, s>>>(in, Bt, out, bias, Bn, H, W, Cin, Cout,
                                      chanMajor);
}

extern "C" void kernel_launch(void* const* d_in, const int* in_sizes, int n_in,
                              void* d_out, int out_size, void* d_ws, size_t ws_size,
                              hipStream_t stream) {
  (void)in_sizes; (void)n_in; (void)out_size; (void)ws_size;
  const float* IN[64];
  for (int i = 0; i < 62; ++i) IN[i] = (const float*)d_in[i];
  const float* xin = IN[0];

  float* wf = (float*)d_ws;
  // ---- workspace layout (float units) ----
  const size_t OFF_XNHWC = 0;                  // 393216
  const size_t OFF_F0    = 393216;             // 8388608  [2,256,256,64]
  const size_t OFF_P0    = OFF_F0 + 8388608;   // 2097152
  const size_t OFF_F1    = OFF_P0 + 2097152;   // 4194304 [2,128,128,128]
  const size_t OFF_P1    = OFF_F1 + 4194304;   // 1048576
  const size_t OFF_F2    = OFF_P1 + 1048576;   // 2097152 [2,64,64,256]
  const size_t OFF_H     = OFF_F2 + 2097152;   // 524288  [2,32,32,256]
  const size_t OFF_MIX   = OFF_H + 524288;     // 524288
  const size_t OFF_GATE  = OFF_MIX + 524288;   // 6144
  const size_t OFF_STATS = OFF_GATE + 6144;    // 1024 (mean@0, var@512)
  const size_t OFF_SA    = OFF_STATS + 1024;   // 16777216 big scratch
  const size_t OFF_SB    = OFF_SA + 16777216;  // 8388608
  const size_t OFF_OUT   = OFF_SB + 8388608;   // 4194304 decoder outs
  const size_t OFF_WBUF  = OFF_OUT + 4194304;  // 1048576 floats (2M bf16)
  const size_t OFF_FIN   = OFF_WBUF + 1048576; // 393216 [131072,3]

  float* xnhwc = wf + OFF_XNHWC;
  float* f0 = wf + OFF_F0;  float* p0 = wf + OFF_P0;
  float* f1 = wf + OFF_F1;  float* p1 = wf + OFF_P1;
  float* f2 = wf + OFF_F2;  float* hb = wf + OFF_H;
  float* mix = wf + OFF_MIX;
  float* gateb = wf + OFF_GATE;
  float* meanb = wf + OFF_STATS;
  float* varb  = wf + OFF_STATS + 512;
  float* SA = wf + OFF_SA;
  float* SB = wf + OFF_SB;
  float* OUTB = wf + OFF_OUT;
  bf16*  WB = (bf16*)(wf + OFF_WBUF);
  float* FINB = wf + OFF_FIN;

  // mamba sub-buffers inside SA
  bf16*  seq_bf = (bf16*)(SA + 0);             // [2048,256] bf16
  float* xz     = SA + 262144;                 // [2048,1024]
  float* xi     = SA + 2359296;                // [2048,512]
  bf16*  xi_bf  = (bf16*)(SA + 3407872);       // [2048,512] bf16
  float* dbl    = SA + 3932160;                // [2048,48]
  bf16*  dt_bf  = (bf16*)(SA + 4030464);       // [2048,32] bf16
  float* delta  = SA + 4063232;                // [2048,512]
  float* yb     = SA + 5111808;                // [2048,512]
  bf16*  y_bf   = (bf16*)(SA + 6160384);       // [2048,512] bf16
  float* etmp   = SA + 6684672;                // [2048,256]

  // ------------------------------ encoder ---------------------------------
  nchw2nhwc<<<gblk(393216), 256, 0, stream>>>(xin, xnhwc, 2, 3, 256, 256);

  struct { const float *w, *b, *g, *bt; int Cin, Cout, H, W; float *out, *pool; }
  enc[3] = {
    { IN[1], IN[2], IN[3], IN[4],   3,  64, 256, 256, f0, p0 },
    { IN[5], IN[6], IN[7], IN[8],  64, 128, 128, 128, f1, p1 },
    { IN[9], IN[10], IN[11], IN[12], 128, 256, 64, 64, f2, hb },
  };
  const float* stage_in = xnhwc;
  for (int s = 0; s < 3; ++s) {
    int cm = (enc[s].Cin % 32 == 0) ? 1 : 0;
    int K = enc[s].Cin * 9, Kp = (K + 31) & ~31, Np = (enc[s].Cout + 15) & ~15;
    prep_bt<<<gblk((size_t)Np * Kp), 256, 0, stream>>>(
        enc[s].w, WB, K, enc[s].Cout, Kp, Np, cm ? 2 : 0, enc[s].Cin);
    run_conv(stage_in, WB, enc[s].out, enc[s].b, 2, enc[s].H, enc[s].W,
             enc[s].Cin, enc[s].Cout, cm, stream);
    int M = 2 * enc[s].H * enc[s].W;
    bn_stats<<<enc[s].Cout, 256, 0, stream>>>(enc[s].out, M, enc[s].Cout, meanb, varb);
    bn_apply<<<gblk((size_t)M * enc[s].Cout), 256, 0, stream>>>(
        enc[s].out, meanb, varb, enc[s].g, enc[s].bt, M, enc[s].Cout, 0);
    maxpool2_nhwc<<<gblk((size_t)M / 4 * enc[s].Cout), 256, 0, stream>>>(
        enc[s].out, enc[s].pool, 2, enc[s].H, enc[s].W, enc[s].Cout);
    stage_in = enc[s].pool;
  }

  // --------------------------- mamba mixture ------------------------------
  const int BL = 2048, DI = 512;
  gate_softmax<<<gblk(BL), 256, 0, stream>>>(hb, IN[46], IN[47], gateb, BL, 256);
  cvt_bf16<<<gblk((size_t)BL * 256), 256, 0, stream>>>(hb, seq_bf, BL, 256, 256, 0, 256);

  for (int e = 0; e < 3; ++e) {
    int base = 13 + e * 11;
    const float *w_in = IN[base + 0], *b_in = IN[base + 1];
    const float *cw = IN[base + 2], *cb = IN[base + 3];
    const float *w_x = IN[base + 4], *w_dt = IN[base + 5], *b_dt = IN[base + 6];
    const float *A_log = IN[base + 7], *Dp = IN[base + 8];
    const float *w_out = IN[base + 9], *b_out = IN[base + 10];

    // xz = seq @ w_in + b_in      [2048,1024]
    prep_bt<<<gblk((size_t)1024 * 256), 256, 0, stream>>>(w_in, WB, 256, 1024, 256, 1024, 1, 0);
    run_gemm(seq_bf, WB, xz, b_in, BL, 1024, 256, 1024, stream);
    // causal depthwise conv + silu -> xi [2048,512]
    dwconv_silu<<<gblk((size_t)BL * DI), 256, 0, stream>>>(xz, cw, cb, xi, 2, 1024, DI);
    cvt_bf16<<<gblk((size_t)BL * DI), 256, 0, stream>>>(xi, xi_bf, BL, DI, DI, 0, DI);
    // dbl = xi @ w_x              [2048,48]
    prep_bt<<<gblk((size_t)48 * 512), 256, 0, stream>>>(w_x, WB, 512, 48, 512, 48, 1, 0);
    run_gemm(xi_bf, WB, dbl, nullptr, BL, 48, 512, 48, stream);
    // delta = softplus(dt @ w_dt + b_dt)   [2048,512]  (K=16 padded to 32)
    cvt_bf16<<<gblk((size_t)BL * 32), 256, 0, stream>>>(dbl, dt_bf, BL, 16, 48, 0, 32);
    prep_bt<<<gblk((size_t)512 * 32), 256, 0, stream>>>(w_dt, WB, 16, 512, 32, 512, 1, 0);
    run_gemm(dt_bf, WB, delta, b_dt, BL, 512, 32, 512, stream);
    softplus_k<<<gblk((size_t)BL * DI), 256, 0, stream>>>(delta, (size_t)BL * DI);
    // selective scan (+D*xi, *silu(z))
    ssm_scan<<<4, 256, 0, stream>>>(delta, dbl, xi, xz, A_log, Dp, yb, 2, 1024, DI);
    // out = y @ w_out + b_out     [2048,256]
    cvt_bf16<<<gblk((size_t)BL * DI), 256, 0, stream>>>(yb, y_bf, BL, DI, DI, 0, DI);
    prep_bt<<<gblk((size_t)256 * 512), 256, 0, stream>>>(w_out, WB, 512, 256, 512, 256, 1, 0);
    run_gemm(y_bf, WB, etmp, b_out, BL, 256, 512, 256, stream);
    gate_accum<<<gblk((size_t)BL * 256), 256, 0, stream>>>(mix, etmp, gateb, e, BL, 256, e == 0);
  }

  // ------------------------- diffusion (100 steps) -------------------------
  diffusion_k<<<512, 256, 0, stream>>>(mix, 2, 256, 100);

  // ------------------------------ decoder ---------------------------------
  struct { const float *w, *b, *g, *bt; int Cup, Cskip, Cout, H, W; const float* skip; }
  dec[3] = {
    { IN[48], IN[49], IN[50], IN[51], 256, 256, 128, 64, 64, f2 },
    { IN[52], IN[53], IN[54], IN[55], 128, 128, 64, 128, 128, f1 },
    { IN[56], IN[57], nullptr, nullptr, 64, 64, 32, 256, 256, f0 },
  };
  const float* dcur = mix;  // [2,32,32,256]
  for (int s = 0; s < 3; ++s) {
    int H = dec[s].H, W = dec[s].W;
    int Cin = dec[s].Cup + dec[s].Cskip;
    size_t Mpos = (size_t)2 * H * W;
    upsample2_nhwc<<<gblk(Mpos * dec[s].Cup), 256, 0, stream>>>(
        dcur, SB, 2, H / 2, W / 2, dec[s].Cup);
    concat_c<<<gblk(Mpos * Cin), 256, 0, stream>>>(SB, dec[s].skip, SA, Mpos,
                                                   dec[s].Cup, dec[s].Cskip);
    int K = Cin * 9, Kp = (K + 31) & ~31, Np = (dec[s].Cout + 15) & ~15;
    prep_bt<<<gblk((size_t)Np * Kp), 256, 0, stream>>>(
        dec[s].w, WB, K, dec[s].Cout, Kp, Np, 2, Cin);
    run_conv(SA, WB, OUTB, dec[s].b, 2, H, W, Cin, dec[s].Cout, 1, stream);
    int M = (int)Mpos;
    if (s < 2) {
      bn_stats<<<dec[s].Cout, 256, 0, stream>>>(OUTB, M, dec[s].Cout, meanb, varb);
      bn_apply<<<gblk((size_t)M * dec[s].Cout), 256, 0, stream>>>(
          OUTB, meanb, varb, dec[s].g, dec[s].bt, M, dec[s].Cout, 0);
    } else {
      act_only<<<gblk((size_t)M * dec[s].Cout), 256, 0, stream>>>(
          OUTB, (size_t)M * dec[s].Cout, 1);
    }
    dcur = OUTB;
  }

  // ------------------------------- final -----------------------------------
  {
    int K = 32 * 9, Kp = (K + 31) & ~31;  // 288
    prep_bt<<<gblk((size_t)16 * Kp), 256, 0, stream>>>(IN[58], WB, K, 3, Kp, 16, 2, 32);
    run_conv(OUTB, WB, FINB, IN[59], 2, 256, 256, 32, 3, 1, stream);
    int M = 131072;
    bn_stats<<<3, 256, 0, stream>>>(FINB, M, 3, meanb, varb);
    bn_apply<<<gblk((size_t)M * 3), 256, 0, stream>>>(FINB, meanb, varb,
                                                      IN[60], IN[61], M, 3, 2);
    nhwc2nchw<<<gblk((size_t)M * 3), 256, 0, stream>>>(FINB, (float*)d_out, 2, 3, 256, 256);
  }
}
